// CausalSelfAttention_90074054131840
// MI455X (gfx1250) — matrix operand order repaired and checked by reading, not compiled, not run on hardware
//
#include <hip/hip_runtime.h>

// ---------------------------------------------------------------------------
// CDNA5 (gfx1250, wave32) causal MQA attention, bf16 WMMA everywhere.
// Pipeline:
//   0) one-shot fp32->bf16 conversion of x and all weights (removes per-tile
//      v_cvt from GEMM inner loops; halves fragment load bandwidth)
//   1) q/k/v projections  (bf16 WMMA, f32 accum)
//   2) fused RoPE+RMSNorm on q,k; V transposed to (b,d,s) bf16
//   3) flash attention (online softmax, causal), bf16 output
//   4) out = Y @ wo^T
// ---------------------------------------------------------------------------

typedef __attribute__((ext_vector_type(16))) __bf16 v16bf;
typedef __attribute__((ext_vector_type(8)))  __bf16 v8bf;
typedef __attribute__((ext_vector_type(4)))  __bf16 v4bf;
typedef __attribute__((ext_vector_type(8)))  float  v8f;

#define WMMA_BF16(a, b, c) \
  __builtin_amdgcn_wmma_f32_16x16x32_bf16(false, (a), false, (b), (short)0, (c), false, false)

// A-fragment (16-bit A 16x32): this lane's 16 K-values at p[0..15].
__device__ __forceinline__ v16bf ldA(const __bf16* p) {
  v8bf lo = *(const v8bf*)p;
  v8bf hi = *(const v8bf*)(p + 8);
  v16bf r;
#pragma unroll
  for (int e = 0; e < 8; ++e) { r[e] = lo[e]; r[8 + e] = hi[e]; }
  return r;
}

__device__ __forceinline__ float redmax16(float v) {
  v = fmaxf(v, __shfl_xor(v, 1));
  v = fmaxf(v, __shfl_xor(v, 2));
  v = fmaxf(v, __shfl_xor(v, 4));
  v = fmaxf(v, __shfl_xor(v, 8));
  return v;
}
__device__ __forceinline__ float redsum16(float v) {
  v += __shfl_xor(v, 1);
  v += __shfl_xor(v, 2);
  v += __shfl_xor(v, 4);
  v += __shfl_xor(v, 8);
  return v;
}

// ---------------------------------------------------------------------------
// One-shot fp32 -> bf16 (n multiple of 1024; 4 elements/thread)
// ---------------------------------------------------------------------------
__global__ __launch_bounds__(256)
void f32_to_bf16_kernel(const float* __restrict__ src, __bf16* __restrict__ dst, int n) {
  int i = (blockIdx.x * 256 + threadIdx.x) * 4;
  if (i < n) {
    float4 v = *(const float4*)(src + i);
    v4bf o;
    o[0] = (__bf16)v.x; o[1] = (__bf16)v.y; o[2] = (__bf16)v.z; o[3] = (__bf16)v.w;
    *(v4bf*)(dst + i) = o;
  }
}

// ---------------------------------------------------------------------------
// C[M,N] = A[M,K] @ W[N,K]^T   (bf16 in, fp32 out; f32 WMMA accumulation)
// grid = (N/64, M/32), block = 32 (one wave).  Wave tile 32x64:
// 2 A-fragments x 4 B-fragments -> 8 WMMAs per 32-wide k-step.
// ---------------------------------------------------------------------------
__global__ __launch_bounds__(32)
void gemm_nt_bf16(const __bf16* __restrict__ A, const __bf16* __restrict__ W,
                  float* __restrict__ C, int M, int N, int K) {
  const int lane = threadIdx.x;
  const int hh = lane >> 4;   // half-wave select
  const int lr = lane & 15;
  const int n0 = blockIdx.x * 64;
  const int m0 = blockIdx.y * 32;
  const __bf16* arow0 = A + (size_t)(m0 + lr) * K;
  const __bf16* arow1 = A + (size_t)(m0 + 16 + lr) * K;

  v8f zz = {};
  v8f acc[2][4];
#pragma unroll
  for (int mi = 0; mi < 2; ++mi)
#pragma unroll
    for (int t = 0; t < 4; ++t) acc[mi][t] = zz;

  for (int k0 = 0; k0 < K; k0 += 32) {
    const int ab = k0 + hh * 16;
    v16bf af0 = ldA(arow0 + ab);
    v16bf af1 = ldA(arow1 + ab);
    v16bf bfr[4];
#pragma unroll
    for (int t = 0; t < 4; ++t)
      bfr[t] = *(const v16bf*)(W + (size_t)(n0 + t * 16 + lr) * K + k0 + hh * 16);
    if (k0 + 512 < K) __builtin_prefetch(arow0 + k0 + 512, 0, 0);
#pragma unroll
    for (int t = 0; t < 4; ++t) {
      acc[0][t] = WMMA_BF16(af0, bfr[t], acc[0][t]);
      acc[1][t] = WMMA_BF16(af1, bfr[t], acc[1][t]);
    }
  }
  // C/D layout: VGPR j, lanes 0-15 -> (M=j, N=lane); lanes 16-31 -> (M=j+8).
#pragma unroll
  for (int mi = 0; mi < 2; ++mi)
#pragma unroll
    for (int t = 0; t < 4; ++t)
#pragma unroll
      for (int j = 0; j < 8; ++j)
        C[(size_t)(m0 + mi * 16 + j + 8 * hh) * N + (n0 + t * 16 + lr)] = acc[mi][t][j];
}

// ---------------------------------------------------------------------------
// Fused RoPE + RMSNorm; src (b,s,h,d) fp32 -> dst (b,h,s,d) bf16.
// One block (128 threads) per (b,s,h) row.  grid.x = B*S*H (h fastest).
// ---------------------------------------------------------------------------
__global__ __launch_bounds__(128)
void rope_rms_kernel(const float* __restrict__ src, __bf16* __restrict__ dst,
                     const float* __restrict__ cosp, const float* __restrict__ sinp,
                     int H) {
  const int S = 2048, D = 128;
  int gid = blockIdx.x;
  int h = gid % H;
  int s = (gid / H) % S;
  int b = gid / (H * S);
  int d = threadIdx.x;
  const float* r = src + (size_t)(b * S + s) * (H * D) + (size_t)h * D;
  int d2 = (d < 64) ? d : d - 64;
  float cs = cosp[s * 64 + d2];
  float sn = sinp[s * 64 + d2];
  float val = (d < 64) ? (r[d] * cs + r[d + 64] * sn)
                       : (-r[d - 64] * sn + r[d] * cs);
  float sq = val * val;
  sq += __shfl_xor(sq, 1);
  sq += __shfl_xor(sq, 2);
  sq += __shfl_xor(sq, 4);
  sq += __shfl_xor(sq, 8);
  sq += __shfl_xor(sq, 16);
  __shared__ float red[4];
  if ((threadIdx.x & 31) == 0) red[threadIdx.x >> 5] = sq;
  __syncthreads();
  float tot = red[0] + red[1] + red[2] + red[3];
  float inv = rsqrtf(tot * (1.0f / 128.0f) + 1.1920928955078125e-07f);
  dst[(((size_t)b * H + h) * S + s) * D + d] = (__bf16)(val * inv);
}

// V (b,s,d) fp32 -> V^T (b,d,s) bf16 so the P@V B-operand is contiguous.
__global__ __launch_bounds__(256)
void v_transpose_kernel(const float* __restrict__ vraw, __bf16* __restrict__ vt) {
  const int S = 2048, D = 128;
  int idx = blockIdx.x * 256 + threadIdx.x;   // over B*S*D = 524288
  int d = idx & (D - 1);
  int s = (idx >> 7) & (S - 1);
  int b = idx >> 18;
  vt[((size_t)b * D + d) * S + s] = (__bf16)vraw[idx];
}

// ---------------------------------------------------------------------------
// Flash attention (causal, MQA: 1 KV head).  grid = (S/64, B*H), block = 128
// (4 waves).  Each wave owns 16 q-rows; online softmax over 32-key tiles.
// Output written directly as bf16 (feeds the final GEMM).
// ---------------------------------------------------------------------------
__global__ __launch_bounds__(128)
void flash_attn_kernel(const __bf16* __restrict__ qn, const __bf16* __restrict__ kn,
                       const __bf16* __restrict__ vt, __bf16* __restrict__ Yb) {
  const int S = 2048, H = 16, D = 128;
  const float scale = 0.088388347648318447f;   // 1/sqrt(128)
  const float NEGINF = -__builtin_inff();
  int lane = threadIdx.x & 31;
  int w = threadIdx.x >> 5;
  int hh = lane >> 4, lr = lane & 15;
  int q0 = blockIdx.x * 64;
  int b = blockIdx.y >> 4, h = blockIdx.y & 15;
  int qr0 = q0 + w * 16;

  // Q A-fragments over D=128 (4 fragments of 16x32), kept in registers.
  const __bf16* qrow = qn + (((size_t)b * H + h) * S + (qr0 + lr)) * D;
  v16bf qf[4];
#pragma unroll
  for (int f = 0; f < 4; ++f) qf[f] = ldA(qrow + f * 32 + hh * 16);

  v8f zz = {};
  v8f o[8];
#pragma unroll
  for (int t = 0; t < 8; ++t) o[t] = zz;
  float mrun[8], lrun[8];
#pragma unroll
  for (int j = 0; j < 8; ++j) { mrun[j] = NEGINF; lrun[j] = 0.f; }

  const __bf16* kb_base = kn + (size_t)b * S * D;
  const __bf16* v_base  = vt + (size_t)b * D * S;
  __shared__ __align__(16) __bf16 pl[4][16][32];   // per-wave P staging

  const int kend = q0 + 64;                        // uniform across block
  for (int kb = 0; kb < kend; kb += 32) {
    // S = Q @ K^T  (two 16x16 key tiles, accumulate over D)
    v8f st[2]; st[0] = zz; st[1] = zz;
#pragma unroll
    for (int t = 0; t < 2; ++t) {
#pragma unroll
      for (int f = 0; f < 4; ++f) {
        const __bf16* kr = kb_base + (size_t)(kb + t * 16 + lr) * D + f * 32 + hh * 16;
        v16bf kf = *(const v16bf*)kr;
        st[t] = WMMA_BF16(qf[f], kf, st[t]);
      }
    }
    // scale + causal mask (register-resident; row/col known per element)
#pragma unroll
    for (int t = 0; t < 2; ++t) {
#pragma unroll
      for (int j = 0; j < 8; ++j) {
        int row = qr0 + j + 8 * hh;
        int col = kb + t * 16 + lr;
        float sv = st[t][j] * scale;
        st[t][j] = (col <= row) ? sv : NEGINF;
      }
    }
    // online softmax: width-16 reductions match the half-wave C layout
    float corr[8];
#pragma unroll
    for (int j = 0; j < 8; ++j) {
      float mn   = redmax16(fmaxf(st[0][j], st[1][j]));
      float mnew = fmaxf(mrun[j], mn);
      corr[j] = __expf(mrun[j] - mnew);
      float p0 = __expf(st[0][j] - mnew);
      float p1 = __expf(st[1][j] - mnew);
      st[0][j] = p0; st[1][j] = p1;
      lrun[j] = lrun[j] * corr[j] + redsum16(p0 + p1);
      mrun[j] = mnew;
    }
#pragma unroll
    for (int t = 0; t < 8; ++t)
#pragma unroll
      for (int j = 0; j < 8; ++j) o[t][j] *= corr[j];

    // C-layout -> A-fragment layout for P via per-wave LDS tile
    __syncthreads();
#pragma unroll
    for (int t = 0; t < 2; ++t)
#pragma unroll
      for (int j = 0; j < 8; ++j)
        pl[w][j + 8 * hh][t * 16 + lr] = (__bf16)st[t][j];
    __syncthreads();
    v16bf pf = ldA(&pl[w][lr][hh * 16]);
    // O += P @ V  (V^T rows are contiguous over keys)
#pragma unroll
    for (int t8 = 0; t8 < 8; ++t8) {
      const __bf16* vr = v_base + (size_t)(t8 * 16 + lr) * S + kb + hh * 16;
      v16bf vf = *(const v16bf*)vr;
      o[t8] = WMMA_BF16(pf, vf, o[t8]);
    }
  }
  // normalize and scatter bf16 to Yb (B,S,H*D)
#pragma unroll
  for (int j = 0; j < 8; ++j) lrun[j] = 1.0f / lrun[j];
#pragma unroll
  for (int t8 = 0; t8 < 8; ++t8)
#pragma unroll
    for (int j = 0; j < 8; ++j) {
      int row = qr0 + j + 8 * hh;
      Yb[((size_t)b * S + row) * 2048 + h * D + t8 * 16 + lr] = (__bf16)(o[t8][j] * lrun[j]);
    }
}

// ---------------------------------------------------------------------------
// Host launcher
// ---------------------------------------------------------------------------
extern "C" void kernel_launch(void* const* d_in, const int* in_sizes, int n_in,
                              void* d_out, int out_size, void* d_ws, size_t ws_size,
                              hipStream_t stream) {
  (void)in_sizes; (void)n_in; (void)out_size; (void)ws_size;
  const float* x    = (const float*)d_in[0];
  const float* cosp = (const float*)d_in[1];
  const float* sinp = (const float*)d_in[2];
  const float* wq   = (const float*)d_in[3];
  const float* wk   = (const float*)d_in[4];
  const float* wv   = (const float*)d_in[5];
  const float* wo   = (const float*)d_in[6];
  float* out = (float*)d_out;

  // Workspace layout (~91 MB):
  char* ws = (char*)d_ws;
  __bf16* xb   = (__bf16*)(ws + 0);          // 16777216 B
  __bf16* wqb  = (__bf16*)(ws + 16777216);   //  8388608 B
  __bf16* wkb  = (__bf16*)(ws + 25165824);   //   524288 B
  __bf16* wvb  = (__bf16*)(ws + 25690112);   //   524288 B
  __bf16* wob  = (__bf16*)(ws + 26214400);   //  8388608 B
  float*  qraw = (float*)(ws + 34603008);    // 33554432 B (reused for Yb)
  float*  kraw = (float*)(ws + 68157440);    //  2097152 B
  float*  vraw = (float*)(ws + 70254592);    //  2097152 B
  __bf16* qn   = (__bf16*)(ws + 72351744);   // 16777216 B
  __bf16* kn   = (__bf16*)(ws + 89128960);   //  1048576 B
  __bf16* vt   = (__bf16*)(ws + 90177536);   //  1048576 B
  __bf16* Yb   = (__bf16*)qraw;              // qraw dead after rope_rms(q)

  // 0) one-shot bf16 conversions (x: 8.4M, wq/wo: 4.2M, wk/wv: 0.26M elems)
  f32_to_bf16_kernel<<<8192, 256, 0, stream>>>(x,  xb,  8388608);
  f32_to_bf16_kernel<<<4096, 256, 0, stream>>>(wq, wqb, 4194304);
  f32_to_bf16_kernel<<<256,  256, 0, stream>>>(wk, wkb, 262144);
  f32_to_bf16_kernel<<<256,  256, 0, stream>>>(wv, wvb, 262144);
  f32_to_bf16_kernel<<<4096, 256, 0, stream>>>(wo, wob, 4194304);

  // 1) projections: M=B*S=4096, K=HID=2048; grid (N/64, M/32)
  gemm_nt_bf16<<<dim3(32, 128), 32, 0, stream>>>(xb, wqb, qraw, 4096, 2048, 2048);
  gemm_nt_bf16<<<dim3(2, 128),  32, 0, stream>>>(xb, wkb, kraw, 4096,  128, 2048);
  gemm_nt_bf16<<<dim3(2, 128),  32, 0, stream>>>(xb, wvb, vraw, 4096,  128, 2048);

  // 2) RoPE + RMSNorm (q: 16 heads, k: 1 head), V transpose/convert
  rope_rms_kernel<<<65536, 128, 0, stream>>>(qraw, qn, cosp, sinp, 16);
  rope_rms_kernel<<<4096,  128, 0, stream>>>(kraw, kn, cosp, sinp, 1);
  v_transpose_kernel<<<2048, 256, 0, stream>>>(vraw, vt);

  // 3) flash attention: grid (S/64, B*H), bf16 output
  flash_attn_kernel<<<dim3(32, 32), 128, 0, stream>>>(qn, kn, vt, Yb);

  // 4) output projection: out = Y @ wo^T
  gemm_nt_bf16<<<dim3(32, 128), 32, 0, stream>>>(Yb, wob, out, 4096, 2048, 2048);
}
